// ACE_ODE_RNN_78941498901281
// MI455X (gfx1250) — matrix operand order repaired and checked by reading, not compiled, not run on hardware
//
#include <hip/hip_runtime.h>
#include <hip/hip_bf16.h>

typedef _Float16 f16;
typedef __attribute__((ext_vector_type(16))) _Float16 v16h;
typedef __attribute__((ext_vector_type(8)))  _Float16 v8h;
typedef __attribute__((ext_vector_type(8)))  float    v8f;

#define NB     16      // batch tile per workgroup
#define DD     32      // D
#define WD     128     // MLP width
#define ND     1024    // D*D
#define NSEQ   48
#define NSTEPS 47
#define NWAVE  8

// ---------------- workspace layout (bytes) ----------------
#define WS_GW2H   0u          // f16 [1024][128]   = 262144
#define WS_ATTWH  262144u     // f16 [1024][1024]  = 2097152
#define WS_H0     2359296u    // f32 [128][32]     = 16384
#define WS_A0     2375680u    // f32 [1024]        = 4096
#define WS_KS     2379776u    // per-wg: 6*(16*32 + 16*1024) f32
#define KS_FLOATS 101376u

// ---------------- LDS layout (bytes) ----------------
enum : unsigned {
  OFF_W0   = 0,       // f16 [256][32]  stacked f_W0|g_W0 (t-col removed)
  OFF_W1   = 16384,   // f16 [256][128] stacked f_W1|g_W1
  OFF_FW2  = 81920,   // f16 [32][128]
  OFF_RW   = 90112,   // f16 [32][64]   stacked [rnn_Wh | rnn_Wx]
  OFF_ATTWX= 94208,   // f16 [1024][32]
  OFF_B0   = 159744,  // f32 256 stacked f_b0|g_b0
  OFF_W0T  = 160768,  // f32 256 stacked t-columns of f_W0|g_W0
  OFF_B1   = 161792,  // f32 256 stacked f_b1|g_b1
  OFF_FB2  = 162816,  // f32 32
  OFF_GB2  = 162944,  // f32 1024
  OFF_RBX  = 167040,  // f32 32
  OFF_ATTBX= 167168,  // f32 1024
  OFF_H    = 171264,  // f32 16*32    state h
  OFF_A    = 173312,  // f32 16*1024  state a
  OFF_HYI  = 238848,  // f32 16*32    stage input
  OFF_AYI  = 240896,  // f32 16*1024  stage input (ALIASED: aT16 f16, out-MLP scratch)
  OFF_INPT = 306432,  // f16 16*32
  OFF_T1   = 307456,  // f16 [16][256] stacked f|g hidden 1
  OFF_T2   = 315648,  // f16 [16][256] stacked f|g hidden 2
  OFF_HX   = 323840,  // f16 [16][64]  stacked [h' | xi]
  OFF_HNT16= 325888,  // f16 16*32     h_new
  OFF_TST  = 326912,  // f32 16
  OFF_DT   = 326976,  // f32 16
  OFF_T0   = 327040,  // f32 16
  SMEM_BYTES = 327168  // <= 327680 (320KB WGP limit)
};

// Tsit5 tables (stage 6 is dead: A_TAB[6]==B_TAB, B_TAB[6]==0)
__constant__ float ATAB[15] = {
  0.161f,
  -0.008480655492356989f, 0.335480655492357f,
  2.8971530571054935f, -6.359448489975075f, 4.3622954328695815f,
  5.325864828439257f, -11.748883564062828f, 7.4955393428898365f, -0.09249506636175525f,
  5.86145544294642f, -12.92096931784711f, 8.159367898576159f, -0.071584973281401f, -0.028269050394068383f
};
__constant__ float CTAB[6] = {0.f, 0.161f, 0.327f, 0.9f, 0.9800255409045097f, 1.0f};
__constant__ float BTAB[6] = {0.09646076681806523f, 0.01f, 0.4798896504144996f,
                              1.379008574103742f, -3.290069515436081f, 2.324710524099774f};

struct KParams {
  const float *ts, *X, *Stat;
  const float *fW0, *fb0, *fW1, *fb1, *fW2, *fb2;
  const float *gW0, *gb0, *gW1, *gb1, *gW2, *gb2;
  const float *injW, *injb;
  const float *rWh, *rWx, *rbx;
  const float *attWh, *attWx, *attbx;
  const float *oW0, *ob0, *oW1, *ob1, *oW2, *ob2;
  float* out;
  char* ws;
};

__device__ inline v8f vz8() { v8f z = {0.f,0.f,0.f,0.f,0.f,0.f,0.f,0.f}; return z; }
__device__ inline float rcp_(float x) { return __builtin_amdgcn_rcpf(x); }
// fast silu via v_rcp_f32 (avoids IEEE div fixup chain)
__device__ inline float silu_(float x) { return x * rcp_(1.f + __expf(-x)); }
// fast tanh: 1 - 2/(e^{2x}+1); exact limits at +-inf with v_rcp
__device__ inline float tanh_(float x) { return 1.f - 2.f * rcp_(__expf(2.f * x) + 1.f); }

__device__ inline v8f wmma16(v16h a, v16h b, v8f c) {
  // v_wmma_f32_16x16x32_f16
  return __builtin_amdgcn_wmma_f32_16x16x32_f16(false, a, false, b, (short)0, c, false, false);
}

// A fragment (16x32 tile of row-major W, leading dim ldw), CDNA5 A layout:
// lanes 0-15: M=m0+lane, K in {k0..k0+7} U {k0+16..k0+23}; lanes 16-31: +8 shift.
__device__ inline v16h ldA16(const f16* W, int ldw, int m0, int k0) {
  int lane = (int)(threadIdx.x & 31u);
  int g = lane >> 4;
  int m = m0 + (lane & 15);
  const f16* p = W + (size_t)m * ldw + k0 + g * 8;
  v8h lo = *(const v8h*)(p);
  v8h hi = *(const v8h*)(p + 16);
  v16h r;
#pragma unroll
  for (int i = 0; i < 8; ++i) { r[i] = lo[i]; r[i + 8] = hi[i]; }
  return r;
}

// B fragment (32x16) from xT layout (xT[b][k], leading dim ldx):
// lanes 0-15: N=lane, K in {k0..k0+7} U {k0+16..k0+23}; lanes 16-31: +8 shift.
__device__ inline v16h ldB16(const f16* xT, int ldx, int k0) {
  int lane = (int)(threadIdx.x & 31u);
  int n = lane & 15, g = lane >> 4;
  const f16* p = xT + (size_t)n * ldx + k0 + g * 8;
  v8h lo = *(const v8h*)(p);
  v8h hi = *(const v8h*)(p + 16);
  v16h r;
#pragma unroll
  for (int i = 0; i < 8; ++i) { r[i] = lo[i]; r[i + 8] = hi[i]; }
  return r;
}

// D = W[m0:m0+16, 0:K] * xT[:, kx0:kx0+K]^T ; separate leading dims for A and B
__device__ inline v8f gemm2(const f16* W, int ldw, int m0,
                            const f16* xT, int ldx, int kx0, int K, v8f acc) {
  for (int k0 = 0; k0 < K; k0 += 32)
    acc = wmma16(ldA16(W, ldw, m0, k0), ldB16(xT, ldx, kx0 + k0), acc);
  return acc;
}

// C/D layout: lane 0-15 -> N=lane, rows m0..m0+7; lanes 16-31 -> rows m0+8..m0+15
__device__ inline void lane_nm(int m0, int& n, int& mb) {
  int lane = (int)(threadIdx.x & 31u);
  n = lane & 15;
  mb = m0 + (lane >> 4) * 8;
}

// -------------------------------------------------------------------------
// Kernel 1: convert streamed weights (g_W2, att_Wh) f32 -> f16 into workspace
// -------------------------------------------------------------------------
__global__ void ace_convert(KParams P) {
  int idx = (int)(blockIdx.x * 256u + threadIdx.x);
  f16* GW2H = (f16*)(P.ws + WS_GW2H);
  f16* AWHH = (f16*)(P.ws + WS_ATTWH);
  if (idx < ND * WD) GW2H[idx] = (f16)P.gW2[idx];
  if (idx < ND * ND) AWHH[idx] = (f16)P.attWh[idx];
}

// -------------------------------------------------------------------------
// Kernel 2: h0 for all 128 elements + shared a0 = corrcoef(h0.T)
// -------------------------------------------------------------------------
__global__ __launch_bounds__(256) void ace_init(KParams P) {
  __shared__ float h0s[128 * DD];
  __shared__ float mean_[DD];
  __shared__ float var_[DD];
  const int tid = (int)threadIdx.x;
  float* H0 = (float*)(P.ws + WS_H0);
  float* A0 = (float*)(P.ws + WS_A0);
  if (tid < 128) {
    const int b = tid;
    float h0raw[DD];
    const float* x0 = P.X + (size_t)b * NSEQ * DD;  // X[b][0][:]
    for (int j = 0; j < DD; ++j) {
      float s = P.rbx[j];
      for (int k = 0; k < DD; ++k) s += P.rWx[j * DD + k] * x0[k];
      h0raw[j] = tanhf(s);
    }
    const float* sb = P.Stat + b * 16;
    for (int j = 0; j < DD; ++j) {
      float s = P.injb[j];
      for (int k = 0; k < DD; ++k) s += P.injW[j * 48 + k] * h0raw[k];
      for (int k = 0; k < 16; ++k) s += P.injW[j * 48 + 32 + k] * sb[k];
      float v = tanhf(s);
      h0s[b * DD + j] = v;
      H0[b * DD + j] = v;
    }
  }
  __syncthreads();
  if (tid < DD) {
    float s = 0.f;
    for (int b = 0; b < 128; ++b) s += h0s[b * DD + tid];
    mean_[tid] = s * (1.f / 128.f);
  }
  __syncthreads();
  if (tid < DD) {
    float mj = mean_[tid], s = 0.f;
    for (int b = 0; b < 128; ++b) { float d = h0s[b * DD + tid] - mj; s += d * d; }
    var_[tid] = s;
  }
  __syncthreads();
  for (int p = tid; p < ND; p += 256) {
    int i = p >> 5, j = p & 31;
    float mi = mean_[i], mj = mean_[j], c = 0.f;
    for (int b = 0; b < 128; ++b)
      c += (h0s[b * DD + i] - mi) * (h0s[b * DD + j] - mj);
    float r = c / sqrtf(var_[i] * var_[j]);
    if (!(fabsf(r) < 1e30f)) r = 0.f;            // nan_to_num
    r = fminf(1.f, fmaxf(-1.f, r));              // corrcoef clips to [-1,1]
    A0[p] = r;
  }
}

// -------------------------------------------------------------------------
// Kernel 3: persistent ODE-RNN. 8 workgroups x 16 batch elements x 8 waves.
// -------------------------------------------------------------------------
__global__ __launch_bounds__(256) void ace_main(KParams P) {
  extern __shared__ char smem[];
  const int tid  = (int)threadIdx.x;
  const int wave = tid >> 5;
  const int bt   = (int)blockIdx.x;
  const int bg   = bt * NB;

  f16* s_W0    = (f16*)(smem + OFF_W0);     // [256][32]  f|g stacked
  f16* s_W1    = (f16*)(smem + OFF_W1);     // [256][128] f|g stacked
  f16* s_fW2   = (f16*)(smem + OFF_FW2);    // [32][128]
  f16* s_rW    = (f16*)(smem + OFF_RW);     // [32][64]   [rWh|rWx]
  f16* s_attWx = (f16*)(smem + OFF_ATTWX);  // [1024][32]
  float* s_b0   = (float*)(smem + OFF_B0);   // [256]
  float* s_w0t  = (float*)(smem + OFF_W0T);  // [256]
  float* s_b1   = (float*)(smem + OFF_B1);   // [256]
  float* s_fb2  = (float*)(smem + OFF_FB2);
  float* s_gb2  = (float*)(smem + OFF_GB2);
  float* s_rbx  = (float*)(smem + OFF_RBX);
  float* s_attbx= (float*)(smem + OFF_ATTBX);
  float* s_h    = (float*)(smem + OFF_H);
  float* s_a    = (float*)(smem + OFF_A);
  float* s_hyi  = (float*)(smem + OFF_HYI);
  float* s_ayi  = (float*)(smem + OFF_AYI);
  f16*   s_aT16 = (f16*)(smem + OFF_AYI);   // alias (used only when a_yi idle)
  f16* s_inpT  = (f16*)(smem + OFF_INPT);   // [16][32]
  f16* s_t1    = (f16*)(smem + OFF_T1);     // [16][256]
  f16* s_t2    = (f16*)(smem + OFF_T2);     // [16][256]
  f16* s_hx    = (f16*)(smem + OFF_HX);     // [16][64]  [h'|xi]
  f16* s_hnT16 = (f16*)(smem + OFF_HNT16);  // [16][32]
  float* s_tst = (float*)(smem + OFF_TST);
  float* s_dt  = (float*)(smem + OFF_DT);
  float* s_t0  = (float*)(smem + OFF_T0);

  const f16* GW2H = (const f16*)(P.ws + WS_GW2H);
  const f16* AWHH = (const f16*)(P.ws + WS_ATTWH);
  const float* H0 = (const float*)(P.ws + WS_H0);
  const float* A0 = (const float*)(P.ws + WS_A0);
  float* ksBase = (float*)(P.ws + WS_KS) + (size_t)bt * KS_FLOATS;
  float* ksh = ksBase;               // [6][16][32]
  float* ksa = ksBase + 6 * NB * DD; // [6][16][1024]

  // ---- stage weights into LDS (f16), f|g stacked ----
  for (int i = tid; i < 2 * WD * DD; i += 256) {
    int m = i >> 5, k = i & 31;
    s_W0[i] = (f16)((m < WD) ? P.fW0[m * 33 + k] : P.gW0[(m - WD) * 33 + k]);
  }
  for (int i = tid; i < WD * WD; i += 256) {
    s_W1[i] = (f16)P.fW1[i];
    s_W1[WD * WD + i] = (f16)P.gW1[i];
  }
  for (int i = tid; i < DD * WD; i += 256) s_fW2[i] = (f16)P.fW2[i];
  for (int i = tid; i < DD * 64; i += 256) {
    int j = i >> 6, k = i & 63;
    s_rW[i] = (f16)((k < DD) ? P.rWh[j * DD + k] : P.rWx[j * DD + (k - DD)]);
  }
  for (int i = tid; i < ND * DD; i += 256) s_attWx[i] = (f16)P.attWx[i];
  for (int i = tid; i < WD; i += 256) {
    s_b0[i] = P.fb0[i];        s_b0[WD + i] = P.gb0[i];
    s_w0t[i] = P.fW0[i*33+32]; s_w0t[WD + i] = P.gW0[i*33+32];
    s_b1[i] = P.fb1[i];        s_b1[WD + i] = P.gb1[i];
  }
  for (int i = tid; i < ND; i += 256) { s_gb2[i] = P.gb2[i]; s_attbx[i] = P.attbx[i]; }
  if (tid < DD) { s_fb2[tid] = P.fb2[tid]; s_rbx[tid] = P.rbx[tid]; }
  // ---- initial state ----
  for (int i = tid; i < NB * DD; i += 256) { int b = i >> 5, j = i & 31; s_h[i] = H0[(bg + b) * DD + j]; }
  for (int i = tid; i < NB * ND; i += 256) s_a[i] = A0[i & (ND - 1)];
  __syncthreads();

#pragma unroll 1
  for (int s = 0; s < NSTEPS; ++s) {
    if (tid < NB) {
      float tp = P.ts[(size_t)(bg + tid) * NSEQ + s];
      float tn = P.ts[(size_t)(bg + tid) * NSEQ + s + 1];
      s_t0[tid] = tp;
      s_dt[tid] = (tn - tp) * 0.5f;
    }
    __syncthreads();

#pragma unroll 1
    for (int sub = 0; sub < 2; ++sub) {
#pragma unroll 1
      for (int st = 0; st < 6; ++st) {
        const int aoff = (st * (st - 1)) >> 1;
        // ---- stage input yi = y + dt * sum_j A[st][j] * ks_j ----
        for (int i = tid; i < NB * ND; i += 256) {
          int b = i >> 10;
          float d = s_dt[b], v = s_a[i];
          for (int j = 0; j < st; ++j) v += d * ATAB[aoff + j] * ksa[j * (NB * ND) + i];
          s_ayi[i] = v;
        }
        for (int i = tid; i < NB * DD; i += 256) {
          int b = i >> 5;
          float d = s_dt[b], v = s_h[i];
          for (int j = 0; j < st; ++j) v += d * ATAB[aoff + j] * ksh[j * (NB * DD) + i];
          s_hyi[i] = v;
        }
        if (tid < NB) s_tst[tid] = s_t0[tid] + CTAB[st] * s_dt[tid];
        __syncthreads();

        // ======== vf(yi) ========
        // fused softmax(a rows) + h_app = attn-row . h  -> inpT (f16, xT layout)
        for (int r = tid; r < NB * DD; r += 256) {
          int b = r >> 5, i = r & 31;
          const float* ar = s_ayi + b * ND + i * DD;
          const float* hr = s_hyi + b * DD;
          float mx = ar[0];
#pragma unroll
          for (int j = 1; j < DD; ++j) mx = fmaxf(mx, ar[j]);
          float sm = 0.f, dot = 0.f;
#pragma unroll
          for (int j = 0; j < DD; ++j) {
            float e = __expf(ar[j] - mx);
            sm += e; dot += e * hr[j];
          }
          s_inpT[b * DD + i] = (f16)(dot * rcp_(sm));
        }
        __syncthreads();
        // layer0 (f|g stacked, M=256): t1 = silu(W0@inp + w0t*t + b0)
        for (int mt = wave; mt < 16; mt += NWAVE) {
          int m0 = mt * 16;
          v8f acc = gemm2(s_W0, DD, m0, s_inpT, DD, 0, DD, vz8());
          int n, mb; lane_nm(m0, n, mb);
          float tv = s_tst[n];
          v8h o;
#pragma unroll
          for (int r2 = 0; r2 < 8; ++r2)
            o[r2] = (f16)silu_(acc[r2] + s_b0[mb + r2] + s_w0t[mb + r2] * tv);
          *(v8h*)(s_t1 + n * 256 + mb) = o;
        }
        __syncthreads();
        // layer1 (f|g stacked, M=256): f-tiles read t1[:,0:128], g-tiles t1[:,128:256]
        for (int mt = wave; mt < 16; mt += NWAVE) {
          int m0 = mt * 16;
          int kx = (mt < 8) ? 0 : WD;
          v8f acc = gemm2(s_W1, WD, m0, s_t1, 256, kx, WD, vz8());
          int n, mb; lane_nm(m0, n, mb);
          v8h o;
#pragma unroll
          for (int r2 = 0; r2 < 8; ++r2)
            o[r2] = (f16)silu_(acc[r2] + s_b1[mb + r2]);
          *(v8h*)(s_t2 + n * 256 + mb) = o;
        }
        __syncthreads();
        // layer2 fused: tiles 0-1 = f_W2 (tanh -> ksh), tiles 2-65 = g_W2 (linear -> ksa)
        for (int mt = wave; mt < 66; mt += NWAVE) {
          if (mt < 2) {
            int m0 = mt * 16;
            v8f acc = gemm2(s_fW2, WD, m0, s_t2, 256, 0, WD, vz8());
            int n, mb; lane_nm(m0, n, mb);
            float* dst = ksh + st * (NB * DD) + n * DD + mb;
#pragma unroll
            for (int r2 = 0; r2 < 8; ++r2) dst[r2] = tanh_(acc[r2] + s_fb2[mb + r2]);
          } else {
            int m0 = (mt - 2) * 16;
            if (mt + NWAVE < 66)
              __builtin_prefetch(GW2H + (size_t)(m0 + NWAVE * 16) * WD, 0, 1);
            v8f acc = gemm2(GW2H, WD, m0, s_t2, 256, WD, WD, vz8());
            int n, mb; lane_nm(m0, n, mb);
            float* dst = ksa + st * (NB * ND) + n * ND + mb;
#pragma unroll
            for (int r2 = 0; r2 < 8; ++r2) dst[r2] = acc[r2] + s_gb2[mb + r2];
          }
        }
        __threadfence_block();   // ks writes (L2) visible to whole workgroup
        __syncthreads();
      } // stages

      // ---- y += dt * sum_j B[j] * ks_j ----
      for (int i = tid; i < NB * ND; i += 256) {
        int b = i >> 10;
        float d = s_dt[b], v = s_a[i];
#pragma unroll
        for (int j = 0; j < 6; ++j) v += d * BTAB[j] * ksa[j * (NB * ND) + i];
        s_a[i] = v;
      }
      for (int i = tid; i < NB * DD; i += 256) {
        int b = i >> 5;
        float d = s_dt[b], v = s_h[i];
#pragma unroll
        for (int j = 0; j < 6; ++j) v += d * BTAB[j] * ksh[j * (NB * DD) + i];
        s_h[i] = v;
      }
      if (sub == 0 && tid < NB) s_t0[tid] += s_dt[tid];
      __syncthreads();
    } // substeps

    // ---- RNN cell: h_new = tanh([rWh|rWx] @ [h';xi] + rbx), single K=64 GEMM ----
    for (int i = tid; i < NB * 64; i += 256) {
      int b = i >> 6, k = i & 63;
      s_hx[i] = (k < DD) ? (f16)s_h[b * DD + k]
                         : (f16)P.X[((size_t)(bg + b) * NSEQ + (s + 1)) * DD + (k - DD)];
    }
    for (int i = tid; i < NB * ND; i += 256) s_aT16[i] = (f16)s_a[i];  // a_yi idle
    __syncthreads();
    if (wave < 2) {
      int m0 = wave * 16;
      v8f acc = gemm2(s_rW, 64, m0, s_hx, 64, 0, 64, vz8());
      int n, mb; lane_nm(m0, n, mb);
#pragma unroll
      for (int r2 = 0; r2 < 8; ++r2) {
        float v = tanh_(acc[r2] + s_rbx[mb + r2]);
        s_h[n * DD + mb + r2]     = v;
        s_hnT16[n * DD + mb + r2] = (f16)v;
      }
    }
    __syncthreads();
    // ---- attention cell: a_new = tanh(attWh@a' + attWx@h_new + attbx) ----
    for (int mt = wave; mt < 64; mt += NWAVE) {
      int m0 = mt * 16;
      if (mt + NWAVE < 64)
        __builtin_prefetch(AWHH + (size_t)(m0 + NWAVE * 16) * ND, 0, 1);
      v8f acc = gemm2(AWHH, ND, m0, s_aT16, ND, 0, ND, vz8());
      acc = wmma16(ldA16(s_attWx, DD, m0, 0), ldB16(s_hnT16, DD, 0), acc);
      int n, mb; lane_nm(m0, n, mb);
      float* dst = s_a + n * ND + mb;
#pragma unroll
      for (int r2 = 0; r2 < 8; ++r2) dst[r2] = tanh_(acc[r2] + s_attbx[mb + r2]);
    }
    __syncthreads();
  } // time steps

  // ---- output MLP (tanh hidden, linear out), scratch in idle a_yi region ----
  float* z0 = s_ayi;            // 16*128
  float* z1 = s_ayi + NB * WD;  // 16*128
  for (int i = tid; i < NB * WD; i += 256) {
    int b = i >> 7, m = i & 127;
    float sv = P.ob0[m];
    const float* hb = s_h + b * DD;
#pragma unroll
    for (int k = 0; k < DD; ++k) sv += P.oW0[m * DD + k] * hb[k];
    z0[i] = tanh_(sv);
  }
  __syncthreads();
  for (int i = tid; i < NB * WD; i += 256) {
    int b = i >> 7, m = i & 127;
    float sv = P.ob1[m];
    const float* zb = z0 + b * WD;
    for (int k = 0; k < WD; ++k) sv += P.oW1[m * WD + k] * zb[k];
    z1[i] = tanh_(sv);
  }
  __syncthreads();
  for (int i = tid; i < NB * 8; i += 256) {
    int b = i >> 3, o = i & 7;
    float sv = P.ob2[o];
    const float* zb = z1 + b * WD;
    for (int k = 0; k < WD; ++k) sv += P.oW2[o * WD + k] * zb[k];
    P.out[(size_t)(bg + b) * 8 + o] = sv;
  }
}

extern "C" void kernel_launch(void* const* d_in, const int* in_sizes, int n_in,
                              void* d_out, int out_size, void* d_ws, size_t ws_size,
                              hipStream_t stream) {
  (void)in_sizes; (void)n_in; (void)out_size; (void)ws_size;
  KParams P;
  P.ts    = (const float*)d_in[0];
  P.X     = (const float*)d_in[1];
  P.Stat  = (const float*)d_in[2];
  P.fW0   = (const float*)d_in[3];
  P.fb0   = (const float*)d_in[4];
  P.fW1   = (const float*)d_in[5];
  P.fb1   = (const float*)d_in[6];
  P.fW2   = (const float*)d_in[7];
  P.fb2   = (const float*)d_in[8];
  P.gW0   = (const float*)d_in[9];
  P.gb0   = (const float*)d_in[10];
  P.gW1   = (const float*)d_in[11];
  P.gb1   = (const float*)d_in[12];
  P.gW2   = (const float*)d_in[13];
  P.gb2   = (const float*)d_in[14];
  P.injW  = (const float*)d_in[15];
  P.injb  = (const float*)d_in[16];
  P.rWh   = (const float*)d_in[17];
  P.rWx   = (const float*)d_in[18];
  P.rbx   = (const float*)d_in[19];
  P.attWh = (const float*)d_in[20];
  P.attWx = (const float*)d_in[21];
  P.attbx = (const float*)d_in[22];
  P.oW0   = (const float*)d_in[23];
  P.ob0   = (const float*)d_in[24];
  P.oW1   = (const float*)d_in[25];
  P.ob1   = (const float*)d_in[26];
  P.oW2   = (const float*)d_in[27];
  P.ob2   = (const float*)d_in[28];
  P.out   = (float*)d_out;
  P.ws    = (char*)d_ws;

  // allow >64KB dynamic LDS (gfx1250: 320KB/WGP); benign if ignored
  (void)hipFuncSetAttribute((const void*)ace_main,
                            hipFuncAttributeMaxDynamicSharedMemorySize,
                            (int)SMEM_BYTES);

  ace_convert<<<(ND * ND + 255) / 256, 256, 0, stream>>>(P);
  ace_init<<<1, 256, 0, stream>>>(P);
  ace_main<<<8, 256, SMEM_BYTES, stream>>>(P);
}
